// Similar_35828617183626
// MI455X (gfx1250) — matrix operand order repaired and checked by reading, not compiled, not run on hardware
//
#include <hip/hip_runtime.h>

// ---------------- types / constants ----------------
typedef __attribute__((ext_vector_type(16))) __bf16 bf16x16;
typedef __attribute__((ext_vector_type(8)))  __bf16 bf16x8;
typedef __attribute__((ext_vector_type(8)))  float  f32x8;
typedef __attribute__((ext_vector_type(4)))  unsigned uint32x4;
typedef __attribute__((ext_vector_type(8)))  int  int32x8;
typedef __attribute__((ext_vector_type(4)))  int  int32x4;

#define NB   8
#define NC   256
#define NPIX 16384          // H*W = 128*128
#define TN   64             // pixels per tile
#define XSTR (NC + 8)       // 264 bf16 (528B, 16B-aligned stride)
#define KSTR (TN + 8)       // 72 bf16 (144B, 16B-aligned stride)
#define KSCALE 0.0078125f   // 1/sqrt(16384)

#define WMMA_BF16(a, b, c) \
  __builtin_amdgcn_wmma_f32_16x16x32_bf16(false, (a), false, (b), (short)0, (c), false, false)

__device__ __forceinline__ f32x8 zero8() {
  f32x8 z;
#pragma unroll
  for (int i = 0; i < 8; ++i) z[i] = 0.0f;
  return z;
}

// A fragment (16x32, M x K) from row-major [M][K] matrix, per CDNA5 layout:
// lanes 0-15: M=lane, K = {0..7, 16..23}; lanes 16-31: M=lane-16, K = {8..15, 24..31}
__device__ __forceinline__ bf16x16 load_a_rm(const __bf16* __restrict__ base, int ld,
                                             int row0, int col0, int lane) {
  int m  = lane & 15;
  int kb = (lane >> 4) << 3;  // 0 or 8
  const __bf16* p = base + (row0 + m) * ld + col0 + kb;
  bf16x8 lo = *(const bf16x8*)(p);        // K = kb+0..7
  bf16x8 hi = *(const bf16x8*)(p + 16);   // K = kb+16..23
  return __builtin_shufflevector(lo, hi, 0,1,2,3,4,5,6,7,8,9,10,11,12,13,14,15);
}

// B fragment (32x16, K x N) from "n-major" storage Bt[n][k] (k contiguous):
// lanes 0-15: N=lane, K = k0+{0..7, 16..23}; lanes 16-31: N=lane-16, K = k0+{8..15, 24..31}
__device__ __forceinline__ bf16x16 load_b_nm(const __bf16* __restrict__ base, int ld,
                                             int n0, int k0, int lane) {
  int n  = lane & 15;
  int kk = k0 + ((lane >> 4) << 3);
  const __bf16* p = base + (n0 + n) * ld + kk;
  bf16x8 lo = *(const bf16x8*)(p);
  bf16x8 hi = *(const bf16x8*)(p + 16);
  return __builtin_shufflevector(lo, hi, 0,1,2,3,4,5,6,7,8,9,10,11,12,13,14,15);
}

__device__ __forceinline__ unsigned pack_bf2(float lo, float hi) {
  __bf16 l = (__bf16)lo, h = (__bf16)hi;
  unsigned short ul = __builtin_bit_cast(unsigned short, l);
  unsigned short uh = __builtin_bit_cast(unsigned short, h);
  return (unsigned)ul | ((unsigned)uh << 16);
}

// Load a [NC x TN] fp32 tile of x and store transposed bf16 Xt[p][c] in LDS.
__device__ __forceinline__ void load_x_tile(const float* __restrict__ x, int b, int p0,
                                            __bf16 (*Xt)[XSTR], int tid) {
  int pairc = (tid & 127) * 2;
  int ph    = (tid >> 7) * 32;
  const float* r0 = x + ((size_t)(b * NC + pairc)) * NPIX + p0 + ph;
  const float* r1 = r0 + NPIX;
#pragma unroll
  for (int i = 0; i < 32; i += 4) {
    float4 a0 = *(const float4*)(r0 + i);
    float4 a1 = *(const float4*)(r1 + i);
    int p = ph + i;
    *(unsigned*)&Xt[p + 0][pairc] = pack_bf2(a0.x, a1.x);
    *(unsigned*)&Xt[p + 1][pairc] = pack_bf2(a0.y, a1.y);
    *(unsigned*)&Xt[p + 2][pairc] = pack_bf2(a0.z, a1.z);
    *(unsigned*)&Xt[p + 3][pairc] = pack_bf2(a0.w, a1.w);
  }
}

// ---------------- TDM: DMA a [256 x 32] bf16 k-panel of a row-major [256][256]
// matrix into LDS (packed, ld=32). D# per CDNA5 ISA ch.8:
//   group0: count=1, lds_addr, global_addr(57b), type=2
//   group1: data_size=1(2B), tensor_dim0=256, tensor_dim1=256, tile=32x256,
//           tensor_dim0_stride=256
__device__ __forceinline__ void tdm_load_panel(const __bf16* gsrc, unsigned lds_off) {
  unsigned long long ga = (unsigned long long)(uintptr_t)gsrc;
  uint32x4 g0 = { 1u,                                     // count=1, user mode
                  lds_off,                                // lds_addr
                  (unsigned)(ga & 0xffffffffu),           // global_addr[31:0]
                  (unsigned)((ga >> 32) & 0x01ffffffu) | (2u << 30) }; // [56:32] | type=2
  int32x8 g1 = { (int)(1u << 16),        // data_size=1 (2 bytes), mask=0
                 (int)(256u << 16),      // tensor_dim0[15:0]=256 (bits 63:48 of g1)
                 (int)(256u << 16),      // tensor_dim0[31:16]=0, tensor_dim1[15:0]=256
                 (int)(32u  << 16),      // tensor_dim1[31:16]=0, tile_dim0=32
                 (int)256,               // tile_dim1=256, tile_dim2=0
                 (int)256,               // tensor_dim0_stride[31:0]=256
                 0, 0 };                 // stride hi / dim1_stride = 0
  int32x4 z4 = { 0, 0, 0, 0 };
#if __clang_major__ >= 23
  int32x8 z8 = { 0, 0, 0, 0, 0, 0, 0, 0 };
  __builtin_amdgcn_tensor_load_to_lds(g0, g1, z4, z4, z8, 0);
#else
  __builtin_amdgcn_tensor_load_to_lds(g0, g1, z4, z4, 0);
#endif
}

// 256x64 GEMM: A = row-major [256][256] bf16 in GLOBAL memory, streamed through
// LDS k-panels by the TDM (double-buffered, s_wait_tensorcnt); B = n-major LDS
// tile (ld=XSTR). Wave w computes output rows [32w, 32w+32) x 64 pixels.
__device__ __forceinline__ void proj_gemm_tdm(const __bf16* __restrict__ Ag,
                                              const __bf16* __restrict__ Bt,
                                              __bf16 (*Wp)[32],   // [2*256][32] dbl buf
                                              int w, int lane, f32x8 (&acc)[2][4]) {
#pragma unroll
  for (int rt = 0; rt < 2; ++rt)
#pragma unroll
    for (int ct = 0; ct < 4; ++ct) acc[rt][ct] = zero8();

  if (w == 0) tdm_load_panel(Ag, (unsigned)(uintptr_t)&Wp[0][0]);

  for (int kt = 0; kt < 8; ++kt) {
    __syncthreads();   // previous compute done -> safe to overwrite idle buffer
    if (w == 0) {
      if (kt < 7) {
        tdm_load_panel(Ag + (kt + 1) * 32, (unsigned)(uintptr_t)&Wp[((kt + 1) & 1) * NC][0]);
        __builtin_amdgcn_s_wait_tensorcnt(1);   // panel kt landed, kt+1 in flight
      } else {
        __builtin_amdgcn_s_wait_tensorcnt(0);
      }
    }
    __syncthreads();   // panel kt visible to all waves
    const __bf16* P = &Wp[(kt & 1) * NC][0];
    bf16x16 a0 = load_a_rm(P, 32, w * 32,      0, lane);
    bf16x16 a1 = load_a_rm(P, 32, w * 32 + 16, 0, lane);
    bf16x16 b0 = load_b_nm(Bt, XSTR,  0, kt * 32, lane);
    bf16x16 b1 = load_b_nm(Bt, XSTR, 16, kt * 32, lane);
    bf16x16 b2 = load_b_nm(Bt, XSTR, 32, kt * 32, lane);
    bf16x16 b3 = load_b_nm(Bt, XSTR, 48, kt * 32, lane);
    acc[0][0] = WMMA_BF16(a0, b0, acc[0][0]);
    acc[0][1] = WMMA_BF16(a0, b1, acc[0][1]);
    acc[0][2] = WMMA_BF16(a0, b2, acc[0][2]);
    acc[0][3] = WMMA_BF16(a0, b3, acc[0][3]);
    acc[1][0] = WMMA_BF16(a1, b0, acc[1][0]);
    acc[1][1] = WMMA_BF16(a1, b1, acc[1][1]);
    acc[1][2] = WMMA_BF16(a1, b2, acc[1][2]);
    acc[1][3] = WMMA_BF16(a1, b3, acc[1][3]);
  }
}

// ---------------- kernel 0: zero kv accumulator + weights -> bf16 ----------------
__global__ __launch_bounds__(256) void k_prep(const float* __restrict__ wq,
                                              const float* __restrict__ wk,
                                              const float* __restrict__ wv,
                                              __bf16* __restrict__ wqb,
                                              __bf16* __restrict__ wkb,
                                              __bf16* __restrict__ wvb,
                                              float* __restrict__ kv) {
  int i = blockIdx.x * 256 + threadIdx.x;
  if (i < NC * NC) {
    wqb[i] = (__bf16)wq[i];
    wkb[i] = (__bf16)wk[i];
    wvb[i] = (__bf16)wv[i];
  }
  if (i < NB * NC * NC) kv[i] = 0.0f;
}

// ---------------- kernel 1: K/V projection + partial kv accumulation ----------------
__global__ __launch_bounds__(256) void k_proj_kv(const float* __restrict__ x,
                                                 const __bf16* __restrict__ Wk,
                                                 const float* __restrict__ bk,
                                                 const __bf16* __restrict__ Wv,
                                                 const float* __restrict__ bv,
                                                 float* __restrict__ kv) {
  __shared__ __bf16 Xt[TN][XSTR];     // x tile, transposed [p][c]
  __shared__ __bf16 Kt[NC][KSTR];     // scaled K tile, [c][p]
  __shared__ __bf16 Vt[NC][KSTR];     // V tile, [C'][p]
  __shared__ __bf16 Wp[2 * NC][32];   // TDM double-buffered weight k-panels
  __shared__ float  bkS[NC], bvS[NC];
  const int tid = threadIdx.x, lane = tid & 31, w = tid >> 5;
  const int b = blockIdx.y, p0 = blockIdx.x * TN;

  bkS[tid] = bk[tid];
  bvS[tid] = bv[tid];
  load_x_tile(x, b, p0, Xt, tid);
  __syncthreads();

  f32x8 acc[2][4];
  // K = Wk*X + bk, scaled by 1/sqrt(HW)
  proj_gemm_tdm(Wk, &Xt[0][0], Wp, w, lane, acc);
#pragma unroll
  for (int rt = 0; rt < 2; ++rt)
#pragma unroll
    for (int ct = 0; ct < 4; ++ct)
#pragma unroll
      for (int r = 0; r < 8; ++r) {
        int o = w * 32 + rt * 16 + ((lane >> 4) << 3) + r;
        int p = ct * 16 + (lane & 15);
        Kt[o][p] = (__bf16)((acc[rt][ct][r] + bkS[o]) * KSCALE);
      }
  // V = Wv*X + bv
  proj_gemm_tdm(Wv, &Xt[0][0], Wp, w, lane, acc);
#pragma unroll
  for (int rt = 0; rt < 2; ++rt)
#pragma unroll
    for (int ct = 0; ct < 4; ++ct)
#pragma unroll
      for (int r = 0; r < 8; ++r) {
        int o = w * 32 + rt * 16 + ((lane >> 4) << 3) + r;
        int p = ct * 16 + (lane & 15);
        Vt[o][p] = (__bf16)(acc[rt][ct][r] + bvS[o]);
      }
  __syncthreads();

  // kv[c][C'] += sum_p Kt[c][p] * Vt[C'][p]  (A = Kt row-major, B = Vt n-major, K=64)
  bf16x16 ka[2][2];
#pragma unroll
  for (int i = 0; i < 2; ++i)
#pragma unroll
    for (int ks = 0; ks < 2; ++ks)
      ka[i][ks] = load_a_rm(&Kt[0][0], KSTR, w * 32 + i * 16, ks * 32, lane);

  float* kvb = kv + (size_t)b * NC * NC;
  for (int nt = 0; nt < 16; ++nt) {
    bf16x16 vb0 = load_b_nm(&Vt[0][0], KSTR, nt * 16, 0, lane);
    bf16x16 vb1 = load_b_nm(&Vt[0][0], KSTR, nt * 16, 32, lane);
    f32x8 c0 = zero8(), c1 = zero8();
    c0 = WMMA_BF16(ka[0][0], vb0, c0);
    c0 = WMMA_BF16(ka[0][1], vb1, c0);
    c1 = WMMA_BF16(ka[1][0], vb0, c1);
    c1 = WMMA_BF16(ka[1][1], vb1, c1);
    int n  = nt * 16 + (lane & 15);
    int mb = w * 32 + ((lane >> 4) << 3);
#pragma unroll
    for (int r = 0; r < 8; ++r) {
      __hip_atomic_fetch_add(&kvb[(mb + r) * NC + n],      c0[r],
                             __ATOMIC_RELAXED, __HIP_MEMORY_SCOPE_AGENT);
      __hip_atomic_fetch_add(&kvb[(mb + 16 + r) * NC + n], c1[r],
                             __ATOMIC_RELAXED, __HIP_MEMORY_SCOPE_AGENT);
    }
  }
}

// ---------------- kernel 1.5: kvT[b][C'][c] = bf16(kv[b][c][C']) ----------------
__global__ __launch_bounds__(256) void k_kvt(const float* __restrict__ kv,
                                             __bf16* __restrict__ kvT) {
  int i = blockIdx.x * 256 + threadIdx.x;
  if (i < NB * NC * NC) {
    int b = i >> 16, r = i & 65535, n = r >> 8, m = r & 255;
    kvT[i] = (__bf16)kv[(b << 16) + (m << 8) + n];
  }
}

// ---------------- kernel 2: Q projection + softmax + out = kvT * q_ ----------------
__global__ __launch_bounds__(256) void k_attn_out(const float* __restrict__ x,
                                                  const __bf16* __restrict__ Wq,
                                                  const float* __restrict__ bq,
                                                  const __bf16* __restrict__ kvT,
                                                  float* __restrict__ out) {
  __shared__ __bf16 Xt[TN][XSTR];     // x tile; reused as q_ tile after softmax
  __shared__ float  Qf[NC][68];       // fp32 Q tile [c][p]
  __shared__ float  red[4][TN];
  __shared__ __bf16 Wp[2 * NC][32];   // TDM double-buffered A-panels
  __shared__ float  bqS[NC];
  const int tid = threadIdx.x, lane = tid & 31, w = tid >> 5;
  const int b = blockIdx.y, p0 = blockIdx.x * TN;

  bqS[tid] = bq[tid];
  load_x_tile(x, b, p0, Xt, tid);
  __syncthreads();

  f32x8 acc[2][4];
  proj_gemm_tdm(Wq, &Xt[0][0], Wp, w, lane, acc);
#pragma unroll
  for (int rt = 0; rt < 2; ++rt)
#pragma unroll
    for (int ct = 0; ct < 4; ++ct)
#pragma unroll
      for (int r = 0; r < 8; ++r) {
        int o = w * 32 + rt * 16 + ((lane >> 4) << 3) + r;
        int p = ct * 16 + (lane & 15);
        Qf[o][p] = acc[rt][ct][r] + bqS[o];
      }
  __syncthreads();

  // softmax over channels, x256 scale; thread group g covers channels [64g,64g+64)
  {
    int p = tid & 63, g = tid >> 6;
    float mx = -3.402823466e38f;
    for (int c = g * 64; c < g * 64 + 64; ++c) mx = fmaxf(mx, Qf[c][p]);
    red[g][p] = mx;
    __syncthreads();
    float gm = fmaxf(fmaxf(red[0][p], red[1][p]), fmaxf(red[2][p], red[3][p]));
    __syncthreads();
    float s = 0.0f;
    for (int c = g * 64; c < g * 64 + 64; ++c) s += __expf(Qf[c][p] - gm);
    red[g][p] = s;
    __syncthreads();
    float inv = 256.0f / (red[0][p] + red[1][p] + red[2][p] + red[3][p]);
    for (int c = g * 64; c < g * 64 + 64; ++c)
      Xt[p][c] = (__bf16)(__expf(Qf[c][p] - gm) * inv);   // q_ tile, n-major
  }
  __syncthreads();

  // out[C'][p] = sum_c kvT[C'][c] * q_[c][p]
  proj_gemm_tdm(kvT + (size_t)b * NC * NC, &Xt[0][0], Wp, w, lane, acc);
  float* ob = out + ((size_t)b * NC) * NPIX + p0;
#pragma unroll
  for (int rt = 0; rt < 2; ++rt)
#pragma unroll
    for (int ct = 0; ct < 4; ++ct)
#pragma unroll
      for (int r = 0; r < 8; ++r) {
        int o = w * 32 + rt * 16 + ((lane >> 4) << 3) + r;
        int p = ct * 16 + (lane & 15);
        ob[(size_t)o * NPIX + p] = acc[rt][ct][r];
      }
}

// ---------------- host launcher ----------------
extern "C" void kernel_launch(void* const* d_in, const int* in_sizes, int n_in,
                              void* d_out, int out_size, void* d_ws, size_t ws_size,
                              hipStream_t stream) {
  const float* x  = (const float*)d_in[0];
  const float* wq = (const float*)d_in[1];
  const float* bq = (const float*)d_in[2];
  const float* wk = (const float*)d_in[3];
  const float* bk = (const float*)d_in[4];
  const float* wv = (const float*)d_in[5];
  const float* bv = (const float*)d_in[6];
  float* out = (float*)d_out;

  char* ws = (char*)d_ws;
  __bf16* wqb = (__bf16*)(ws + 0);
  __bf16* wkb = (__bf16*)(ws + 131072);
  __bf16* wvb = (__bf16*)(ws + 262144);
  __bf16* kvT = (__bf16*)(ws + 393216);      // 1 MB
  float*  kv  = (float*) (ws + 1441792);     // 2 MB

  k_prep<<<2048, 256, 0, stream>>>(wq, wk, wv, wqb, wkb, wvb, kv);
  k_proj_kv<<<dim3(NPIX / TN, NB), 256, 0, stream>>>(x, wkb, bk, wvb, bv, kv);
  k_kvt<<<2048, 256, 0, stream>>>(kv, kvT);
  k_attn_out<<<dim3(NPIX / TN, NB), 256, 0, stream>>>(x, wqb, bq, kvT, out);
}
